// EQGATDynamicLocalEdge_88613765251900
// MI455X (gfx1250) — matrix operand order repaired and checked by reading, not compiled, not run on hardware
//
#include <hip/hip_runtime.h>
#include <hip/hip_bf16.h>

// ---------------- problem constants ----------------
#define NG     8      // groups
#define NPG    64     // nodes per group
#define SDIM   64
#define VDIM   16
#define EDIM   16
#define LYRS   5
#define DIN    146    // 2*SDIM + 2 + EDIM
#define KPAD   160    // DIN padded to multiple of 32 for WMMA K-steps
#define HID    64
#define DOUT   97     // SDIM + 2*VDIM + 1
#define NOUT   112    // DOUT padded to multiple of 16
#define EPG    4032   // edges per group = 64*63
#define PI_F   3.14159265358979323846f
#define CUT_F  5.0f

typedef __attribute__((ext_vector_type(16))) _Float16 v16h;
typedef __attribute__((ext_vector_type(8)))  float    v8f;

// A fragment (16x32 f16, M rows per lane): lanes 0-15 hold K 0..15,
// lanes 16-31 hold K 16..31 (contiguous, matching load_b_frag).
__device__ __forceinline__ v16h load_a_frag(const _Float16* A, int lda,
                                            int mbase, int kbase, int lane) {
  const int m   = mbase + (lane & 15);
  const int khi = (lane >> 4) << 4;          // 0 or 16
  const _Float16* row = A + m * lda + kbase;
  v16h a;
#pragma unroll
  for (int e = 0; e < 16; ++e) {
    const int k = e + khi;
    a[e] = row[k];
  }
  return a;
}

// B fragment (32x16 f16, N cols across lanes): lanes 0-15 hold K 0..15,
// lanes 16-31 hold K 16..31 of the 32-chunk.
__device__ __forceinline__ v16h load_b_frag(const _Float16* B, int ldb,
                                            int kbase, int nbase, int lane) {
  const int n    = nbase + (lane & 15);
  const int koff = (lane >> 4) << 4;         // 0 or 16
  const _Float16* col = B + (kbase + koff) * ldb + n;
  v16h b;
#pragma unroll
  for (int e = 0; e < 16; ++e) b[e] = col[e * ldb];
  return b;
}

// Fast transcendentals: v_exp_f32 / v_cos_f32 / v_rcp_f32 instead of libm
// range-reduction ladders (silu inputs are O(1..10), cos args <~ 5 rad on
// masked pairs; unmasked pairs are zeroed by the mask).
__device__ __forceinline__ float fast_exp(float x) {
  return __builtin_amdgcn_exp2f(x * 1.44269504088896340736f);
}
__device__ __forceinline__ float fast_cos(float x) {
  return __builtin_amdgcn_cosf(x * 0.15915494309189533577f);   // cos(2*pi*s)
}
__device__ __forceinline__ float silu_f(float x) {
  return x * __builtin_amdgcn_rcpf(1.0f + fast_exp(-x));
}

__global__ __launch_bounds__(256)
void eqgat_group_kernel(const float* __restrict__ s_in,
                        const float* __restrict__ v_in,
                        const float* __restrict__ p_in,
                        const float* __restrict__ ea_in,
                        const float* __restrict__ ln_g,
                        const float* __restrict__ ln_b,
                        const float* __restrict__ W1g,
                        const float* __restrict__ b1g,
                        const float* __restrict__ W2g,
                        const float* __restrict__ b2g,
                        const float* __restrict__ Wm1g,
                        const float* __restrict__ bm1g,
                        const float* __restrict__ Wm2g,
                        const float* __restrict__ bm2g,
                        const float* __restrict__ Wpreg,
                        const float* __restrict__ bpreg,
                        const float* __restrict__ Wpostg,
                        const float* __restrict__ bpostg,
                        float* __restrict__ Ews,     // workspace: NG*NPG*NPG*EDIM f32
                        float* __restrict__ out) {
  const int g    = blockIdx.x;
  const int tid  = threadIdx.x;
  const int lane = tid & 31;
  const int wid  = tid >> 5;

  // ---------------- LDS (~152 KB) ----------------
  __shared__ float     sS[NPG][SDIM];      // residual s (f32)
  __shared__ _Float16  sSLN[NPG][SDIM];    // layernormed s (f16)
  __shared__ float     sV[NPG][3][VDIM];
  __shared__ float     sVacc[NPG][3][VDIM];
  __shared__ float     sP[NPG][4];
  __shared__ float     sPacc[NPG][4];
  __shared__ float     sPN[NPG][4];
  __shared__ unsigned char sMask[NPG][NPG];
  __shared__ float     sDeg[NPG];
  __shared__ _Float16  sW1[KPAD][HID];
  __shared__ _Float16  sW2[HID][NOUT];
  __shared__ _Float16  sA[NPG][KPAD];      // pairwise MLP input tile (row i, all j)
  __shared__ _Float16  sH[NPG][HID];       // silu(hidden), f16
  __shared__ float     sO[NPG][NOUT];      // MLP output tile f32 (also reused as t1)
  __shared__ float     sF[NPG][EDIM];
  __shared__ float     sD[NPG], sCos[NPG], sEnv[NPG];
  __shared__ float     sRN[NPG][4];
  __shared__ float     sWpost[EDIM][EDIM];
  __shared__ float     sBpost[EDIM];

  float* Eg = Ews + (size_t)g * (NPG * NPG * EDIM);

  // ---------------- setup: load state, mask, deg, E ----------------
  for (int idx = tid; idx < NPG * SDIM; idx += 256)
    sS[idx / SDIM][idx % SDIM] = s_in[((size_t)g * NPG + idx / SDIM) * SDIM + (idx % SDIM)];
  for (int idx = tid; idx < NPG * 3 * VDIM; idx += 256) {
    const int n = idx / 48, r = idx % 48;
    sV[n][r / 16][r % 16] = v_in[((size_t)g * NPG + n) * 48 + r];
  }
  for (int idx = tid; idx < NPG * 3; idx += 256)
    sP[idx / 3][idx % 3] = p_in[((size_t)g * NPG + idx / 3) * 3 + (idx % 3)];
  __syncthreads();

  // static adjacency from initial positions (dist0 < CUT, i != j)
  for (int pr = tid; pr < NPG * NPG; pr += 256) {
    const int i = pr / NPG, j = pr % NPG;
    const float dx = sP[j][0] - sP[i][0];
    const float dy = sP[j][1] - sP[i][1];
    const float dz = sP[j][2] - sP[i][2];
    const float d0 = sqrtf(fmaxf(dx * dx + dy * dy + dz * dz, 1e-12f));
    sMask[i][j] = (i != j && d0 < CUT_F) ? 1 : 0;
  }
  // materialize E block from edge_attr (meshgrid-minus-diagonal order)
  for (int idx = tid; idx < NPG * NPG * EDIM; idx += 256) {
    const int pr = idx / EDIM, c = idx % EDIM;
    const int i = pr / NPG, j = pr % NPG;
    float val = 0.0f;
    if (i != j) {
      const int le = i * 63 + (j > i ? j - 1 : j);
      val = ea_in[((size_t)g * EPG + le) * EDIM + c];
    }
    Eg[idx] = val;
  }
  __threadfence_block();
  __syncthreads();
  if (tid < NPG) {
    float dsum = 0.0f;
    for (int j = 0; j < NPG; ++j) dsum += (float)sMask[tid][j];
    sDeg[tid] = fmaxf(dsum, 1.0f);
  }
  __syncthreads();

  // ---------------- layers ----------------
  for (int L = 0; L < LYRS; ++L) {
    // ---- node phase: LayerNorm, v-normalize, pn, zero accumulators ----
    if (tid < NPG) {
      const int n = tid;
      float mu = 0.0f;
      for (int c = 0; c < SDIM; ++c) mu += sS[n][c];
      mu *= (1.0f / SDIM);
      float var = 0.0f;
      for (int c = 0; c < SDIM; ++c) { const float t = sS[n][c] - mu; var += t * t; }
      var *= (1.0f / SDIM);
      const float rstd = rsqrtf(var + 1e-6f);
      for (int c = 0; c < SDIM; ++c)
        sSLN[n][c] = (_Float16)(((sS[n][c] - mu) * rstd) * ln_g[L * SDIM + c] + ln_b[L * SDIM + c]);
      float vs = 0.0f;
      for (int a = 0; a < 3; ++a)
        for (int k = 0; k < VDIM; ++k) vs += sV[n][a][k] * sV[n][a][k];
      const float inv_vn = 1.0f / sqrtf(vs * (1.0f / VDIM) + 1e-6f);
      for (int a = 0; a < 3; ++a)
        for (int k = 0; k < VDIM; ++k) { sV[n][a][k] *= inv_vn; sVacc[n][a][k] = 0.0f; }
      const float pp = sqrtf(sP[n][0] * sP[n][0] + sP[n][1] * sP[n][1] + sP[n][2] * sP[n][2]);
      const float ip = 1.0f / fmaxf(pp, 1e-20f);
      for (int a = 0; a < 3; ++a) { sPN[n][a] = sP[n][a] * ip; sPacc[n][a] = 0.0f; }
    }
    // stage layer weights as f16 (zero-padded)
    for (int idx = tid; idx < KPAD * HID; idx += 256) {
      const int k = idx / HID, c = idx % HID;
      sW1[k][c] = (_Float16)((k < DIN) ? W1g[((size_t)L * DIN + k) * HID + c] : 0.0f);
    }
    for (int idx = tid; idx < HID * NOUT; idx += 256) {
      const int k = idx / NOUT, c = idx % NOUT;
      sW2[k][c] = (_Float16)((c < DOUT) ? W2g[((size_t)L * HID + k) * DOUT + c] : 0.0f);
    }
    __syncthreads();

    // ---- pairwise MLP: loop rows i, 64 pairs per tile ----
    for (int irow = 0; irow < NPG; ++irow) {
      if (tid < NPG) {
        const int j = tid;
        const float rx = sP[j][0] - sP[irow][0];
        const float ry = sP[j][1] - sP[irow][1];
        const float rz = sP[j][2] - sP[irow][2];
        const float d  = sqrtf(fmaxf(rx * rx + ry * ry + rz * rz, 1e-6f));
        sD[j]   = d;
        sCos[j] = sPN[irow][0] * sPN[j][0] + sPN[irow][1] * sPN[j][1] + sPN[irow][2] * sPN[j][2];
        const float inv1d = __builtin_amdgcn_rcpf(1.0f + d);
        sRN[j][0] = rx * inv1d; sRN[j][1] = ry * inv1d; sRN[j][2] = rz * inv1d;
        sEnv[j] = 0.5f * (fast_cos(PI_F * d * (1.0f / CUT_F)) + 1.0f) * (float)sMask[irow][j];
      }
      __syncthreads();

      // build A tile: [ s_i | s_j | d | cos | E_ij | pad ]
      for (int idx = tid; idx < NPG * KPAD; idx += 256) {
        const int j = idx / KPAD, k = idx % KPAD;
        _Float16 val;
        if      (k < 64)  val = sSLN[irow][k];
        else if (k < 128) val = sSLN[j][k - 64];
        else if (k == 128) val = (_Float16)sD[j];
        else if (k == 129) val = (_Float16)sCos[j];
        else if (k < DIN)  val = (_Float16)Eg[(irow * NPG + j) * EDIM + (k - 130)];
        else               val = (_Float16)0.0f;
        sA[j][k] = val;
      }
      __syncthreads();

      // GEMM1: (64 x 160) x (160 x 64) -> silu -> sH (f16)
      for (int t = wid; t < 16; t += 8) {
        const int mt = t >> 2, nt = t & 3;
        v8f acc = {};
#pragma unroll
        for (int kc = 0; kc < KPAD / 32; ++kc) {
          v16h a = load_a_frag(&sA[0][0], KPAD, mt * 16, kc * 32, lane);
          v16h b = load_b_frag(&sW1[0][0], HID, kc * 32, nt * 16, lane);
          acc = __builtin_amdgcn_wmma_f32_16x16x32_f16(false, a, false, b,
                                                       (short)0, acc, false, false);
        }
        const int nc    = nt * 16 + (lane & 15);
        const int mbase = mt * 16 + ((lane >> 4) << 3);
        const float bias = b1g[L * HID + nc];
#pragma unroll
        for (int r = 0; r < 8; ++r)
          sH[mbase + r][nc] = (_Float16)silu_f(acc[r] + bias);
      }
      __syncthreads();

      // GEMM2: (64 x 64) x (64 x 112) -> sO (f32, +bias)
      for (int t = wid; t < 28; t += 8) {
        const int mt = t / 7, nt = t % 7;
        v8f acc = {};
#pragma unroll
        for (int kc = 0; kc < 2; ++kc) {
          v16h a = load_a_frag(&sH[0][0], HID, mt * 16, kc * 32, lane);
          v16h b = load_b_frag(&sW2[0][0], NOUT, kc * 32, nt * 16, lane);
          acc = __builtin_amdgcn_wmma_f32_16x16x32_f16(false, a, false, b,
                                                       (short)0, acc, false, false);
        }
        const int nc    = nt * 16 + (lane & 15);
        const int mbase = mt * 16 + ((lane >> 4) << 3);
        const float bias = (nc < DOUT) ? b2g[L * DOUT + nc] : 0.0f;
#pragma unroll
        for (int r = 0; r < 8; ++r) sO[mbase + r][nc] = acc[r] + bias;
      }
      __syncthreads();

      // reductions over j for node irow:  gs|gr|gv|gc = o[:, 0:64|64:80|80:96|96]
      if (tid < SDIM) {
        const int c = tid;
        float acc = 0.0f;
        for (int j = 0; j < NPG; ++j) acc += sO[j][c] * sEnv[j];
        sS[irow][c] = (float)sSLN[irow][c] + acc;     // residual base is LN'd s
      } else if (tid < SDIM + 48) {
        const int r = tid - SDIM, a = r / 16, k = r % 16;
        float acc = 0.0f;
        for (int j = 0; j < NPG; ++j) {
          const float e = sEnv[j];
          float m = sRN[j][a] * (sO[j][64 + k] * e);
          if (L > 0) m += sV[j][a][k] * (sO[j][80 + k] * e);
          acc += m;
        }
        sVacc[irow][a][k] = acc;
      } else if (tid < SDIM + 48 + 3) {
        const int a = tid - (SDIM + 48);
        float acc = 0.0f;
        for (int j = 0; j < NPG; ++j) acc += sO[j][96] * sEnv[j] * sRN[j][a];
        sPacc[irow][a] = acc;
      }
      __syncthreads();
    } // irow

    // ---- apply buffered v / p updates ----
    for (int idx = tid; idx < NPG * 3 * VDIM; idx += 256) {
      const int n = idx / 48, r = idx % 48;
      sV[n][r / 16][r % 16] += sVacc[n][r / 16][r % 16] / sDeg[n];
    }
    for (int idx = tid; idx < NPG * 3; idx += 256) {
      const int n = idx / 3, a = idx % 3;
      sP[n][a] += sPacc[n][a] / sDeg[n];
    }
    __syncthreads();

    // ---- node MLP residual (layers 0..L-2) ----
    if (L < LYRS - 1) {
      for (int idx = tid; idx < NPG * SDIM; idx += 256) {
        const int n = idx / SDIM, c = idx % SDIM;
        float acc = bm1g[L * SDIM + c];
        for (int k = 0; k < SDIM; ++k)
          acc += sS[n][k] * Wm1g[((size_t)L * SDIM + k) * SDIM + c];
        sO[n][c] = silu_f(acc);          // t1 staged in sO
      }
      __syncthreads();
      for (int idx = tid; idx < NPG * SDIM; idx += 256) {
        const int n = idx / SDIM, c = idx % SDIM;
        float acc = bm2g[L * SDIM + c];
        for (int k = 0; k < SDIM; ++k)
          acc += sO[n][k] * Wm2g[((size_t)L * SDIM + k) * SDIM + c];
        sS[n][c] += acc;
      }
      __syncthreads();
    }

    // ---- edge feature update: f = s@Wpre+bpre; E = (1-m)E + m((ef+silu(fi+fj))@Wpost+bpost) ----
    for (int idx = tid; idx < NPG * EDIM; idx += 256) {
      const int n = idx / EDIM, c = idx % EDIM;
      float acc = bpreg[L * EDIM + c];
      for (int k = 0; k < SDIM; ++k)
        acc += sS[n][k] * Wpreg[((size_t)L * SDIM + k) * EDIM + c];
      sF[n][c] = acc;
    }
    if (tid < EDIM * EDIM) sWpost[tid / EDIM][tid % EDIM] = Wpostg[L * EDIM * EDIM + tid];
    if (tid < EDIM) sBpost[tid] = bpostg[L * EDIM + tid];
    __syncthreads();
    for (int pr = tid; pr < NPG * NPG; pr += 256) {
      const int i = pr / NPG, j = pr % NPG;
      if (sMask[i][j]) {
        float* ef = &Eg[pr * EDIM];
        float t[EDIM];
#pragma unroll
        for (int c = 0; c < EDIM; ++c) {
          const float x = sF[i][c] + sF[j][c];
          t[c] = ef[c] + silu_f(x);
        }
        float ov[EDIM];
#pragma unroll
        for (int c = 0; c < EDIM; ++c) {
          float acc = sBpost[c];
          for (int k = 0; k < EDIM; ++k) acc += t[k] * sWpost[k][c];
          ov[c] = acc;
        }
#pragma unroll
        for (int c = 0; c < EDIM; ++c) ef[c] = ov[c];
      }
    }
    __threadfence_block();
    __syncthreads();
  } // layers

  // ---------------- write outputs: [s | v | e_out | p] ----------------
  for (int idx = tid; idx < NPG * SDIM; idx += 256)
    out[((size_t)g * NPG + idx / SDIM) * SDIM + (idx % SDIM)] = sS[idx / SDIM][idx % SDIM];
  float* vout = out + 512 * SDIM;                       // 32768
  for (int idx = tid; idx < NPG * 48; idx += 256) {
    const int n = idx / 48, r = idx % 48;
    vout[((size_t)g * NPG + n) * 48 + r] = sV[n][r / 16][r % 16];
  }
  float* eout = out + 512 * SDIM + 512 * 48;            // 57344
  for (int idx = tid; idx < EPG * EDIM; idx += 256) {
    const int le = idx / EDIM, c = idx % EDIM;
    const int i = le / 63, jj = le % 63;
    const int j = jj + (jj >= i ? 1 : 0);
    eout[((size_t)g * EPG + le) * EDIM + c] = Eg[(i * NPG + j) * EDIM + c];
  }
  float* pout = out + 512 * SDIM + 512 * 48 + (size_t)NG * EPG * EDIM;  // 573440
  for (int idx = tid; idx < NPG * 3; idx += 256)
    pout[((size_t)g * NPG + idx / 3) * 3 + (idx % 3)] = sP[idx / 3][idx % 3];
}

extern "C" void kernel_launch(void* const* d_in, const int* in_sizes, int n_in,
                              void* d_out, int out_size, void* d_ws, size_t ws_size,
                              hipStream_t stream) {
  (void)in_sizes; (void)n_in; (void)out_size; (void)ws_size;
  const float* s_in  = (const float*)d_in[0];
  const float* v_in  = (const float*)d_in[1];
  const float* p_in  = (const float*)d_in[2];
  const float* ea_in = (const float*)d_in[3];
  // d_in[4] edge_index_global, d_in[5] batch: structure known analytically, unused
  const float* ln_g  = (const float*)d_in[6];
  const float* ln_b  = (const float*)d_in[7];
  const float* W1    = (const float*)d_in[8];
  const float* b1    = (const float*)d_in[9];
  const float* W2    = (const float*)d_in[10];
  const float* b2    = (const float*)d_in[11];
  const float* Wm1   = (const float*)d_in[12];
  const float* bm1   = (const float*)d_in[13];
  const float* Wm2   = (const float*)d_in[14];
  const float* bm2   = (const float*)d_in[15];
  const float* Wpre  = (const float*)d_in[16];
  const float* bpre  = (const float*)d_in[17];
  const float* Wpost = (const float*)d_in[18];
  const float* bpost = (const float*)d_in[19];

  float* Ews = (float*)d_ws;   // NG*NPG*NPG*EDIM f32 = 2 MB
  float* out = (float*)d_out;

  eqgat_group_kernel<<<dim3(NG), dim3(256), 0, stream>>>(
      s_in, v_in, p_in, ea_in, ln_g, ln_b, W1, b1, W2, b2,
      Wm1, bm1, Wm2, bm2, Wpre, bpre, Wpost, bpost, Ews, out);
}
